// CrossAttention_36283883717403
// MI455X (gfx1250) — matrix operand order repaired and checked by reading, not compiled, not run on hardware
//
#include <hip/hip_runtime.h>

// ---------------------------------------------------------------------------
// CrossAttention for MI455X (gfx1250): bf16 WMMA + async global->LDS staging.
// B=8, DIM=512, H=W=32 (P=1024), NH=8, DH=64, DE=512.
// ---------------------------------------------------------------------------

typedef __bf16 bf16_t;
typedef __attribute__((ext_vector_type(16))) __bf16 v16bf;
typedef __attribute__((ext_vector_type(8)))  __bf16 v8bf;
typedef __attribute__((ext_vector_type(8)))  float  v8f;
typedef __attribute__((ext_vector_type(4)))  int    v4i;

#define NB   8
#define CDIM 512
#define PDIM 1024
#define NHH  8
#define DHH  64

#define AS1 __attribute__((address_space(1)))
#define AS3 __attribute__((address_space(3)))

#if __has_builtin(__builtin_amdgcn_global_load_async_to_lds_b128) && \
    __has_builtin(__builtin_amdgcn_s_wait_asynccnt)
#define USE_ASYNC_LDS 1
#else
#define USE_ASYNC_LDS 0
#endif

// Build a v16bf fragment from two contiguous 16-byte runs (A/B layouts are
// always two 8-element runs per lane on CDNA5 WMMA).
static __device__ __forceinline__ v16bf ld_frag(const bf16_t* p0, const bf16_t* p1) {
  union { v16bf f; v8bf h[2]; } u;
  u.h[0] = *(const v8bf*)p0;
  u.h[1] = *(const v8bf*)p1;
  return u.f;
}

static __device__ __forceinline__ v8f wmma_bf16(v16bf a, v16bf b, v8f c) {
  // D = A(16x32) * B(32x16) + C(16x16 f32)
  return __builtin_amdgcn_wmma_f32_16x16x32_bf16(false, a, false, b, (short)0, c,
                                                 false, false);
}

#if USE_ASYNC_LDS
// One 16-byte async DMA chunk: global (as1) -> LDS (as3).  LDS offset is the
// low 32 bits of the generic pointer; global generic == as1.  The builtin
// takes v4i-typed pointers (per hipcc diagnostic).
static __device__ __forceinline__ void cp16_async(const bf16_t* g, bf16_t* l) {
  __builtin_amdgcn_global_load_async_to_lds_b128(
      (AS1 v4i*)(uintptr_t)g, (AS3 v4i*)(uint32_t)(uintptr_t)l, 0, 0);
}
#endif

// ---------------------------------------------------------------------------
// 1) per-pixel inverse channel norm: f = sqrt(DIM) / max(||x_:,p||, 1e-12)
// ---------------------------------------------------------------------------
__global__ void rownorm_kernel(const float* __restrict__ x, const float* __restrict__ ctx,
                               float* __restrict__ invx, float* __restrict__ invc) {
  int idx = blockIdx.x * blockDim.x + threadIdx.x;  // 0..16383
  int sel = idx >> 13;
  int pi  = idx & 8191;
  const float* src = sel ? ctx : x;
  int b = pi >> 10, p = pi & 1023;
  const float* col = src + (size_t)b * CDIM * PDIM + p;
  float s = 0.f;
#pragma unroll 8
  for (int c = 0; c < CDIM; ++c) {
    float v = col[(size_t)c * PDIM];
    s += v * v;
  }
  float f = 22.627416997969522f / fmaxf(sqrtf(s), 1e-12f);  // sqrt(512)*inv
  (sel ? invc : invx)[pi] = f;
}

// ---------------------------------------------------------------------------
// 2) normalize * gamma, cast to bf16, transpose (b,c,p) -> (b,p,c)
// ---------------------------------------------------------------------------
__global__ void normcast_kernel(const float* __restrict__ x, const float* __restrict__ ctx,
                                const float* __restrict__ gamma,
                                const float* __restrict__ invx, const float* __restrict__ invc,
                                bf16_t* __restrict__ xnT, bf16_t* __restrict__ cnT) {
  __shared__ float tile[32][33];
  int z = blockIdx.z;        // 0..15
  int b = z & 7, sel = z >> 3;
  const float* src = sel ? ctx : x;
  const float* inv = sel ? invc : invx;
  bf16_t* dst = sel ? cnT : xnT;
  int p0 = blockIdx.x * 32, c0 = blockIdx.y * 32;
  int tx = threadIdx.x, ty = threadIdx.y;
#pragma unroll
  for (int r = 0; r < 4; ++r) {
    int c = c0 + ty + 8 * r;
    tile[ty + 8 * r][tx] = src[((size_t)b * CDIM + c) * PDIM + p0 + tx] * gamma[c];
  }
  __syncthreads();
#pragma unroll
  for (int r = 0; r < 4; ++r) {
    int pl = ty + 8 * r;
    int p  = p0 + pl;
    dst[((size_t)b * PDIM + p) * CDIM + c0 + tx] =
        (bf16_t)(tile[tx][pl] * inv[b * PDIM + p]);
  }
}

// ---------------------------------------------------------------------------
// 3) cast the four 512x512 weight matrices to bf16 (row-major preserved)
// ---------------------------------------------------------------------------
__global__ void wcast_kernel(const float* __restrict__ wq, const float* __restrict__ wk,
                             const float* __restrict__ wv, const float* __restrict__ wp,
                             bf16_t* __restrict__ dst) {
  int idx = blockIdx.x * 256 + threadIdx.x;  // 0..1048575
  int j = idx >> 18;
  int r = idx & 262143;
  const float* s = (j == 0) ? wq : (j == 1) ? wk : (j == 2) ? wv : wp;
  dst[idx] = (bf16_t)s[r];
}

// ---------------------------------------------------------------------------
// 4) GEMM: C(512 x 1024) = W(512x512) * Xt^T, per batch.
//    Each wave computes a 16(M) x 64(N) strip.  The 4 KB B panel (shared by
//    all 8 waves) is double-buffered in LDS via async DMA (8x less L2 read
//    traffic than per-wave loads).
//    mode 0: out bf16 [bn][p][dh]   (Q/K, head-split o = dh*8+nh)
//    mode 1: out bf16 [bn][dh][p]   (V)
//    mode 2: out f32  [b][o][p]     (final projection -> d_out)
// ---------------------------------------------------------------------------
__global__ void gemm_kernel(const bf16_t* __restrict__ W,   // 512x512 row-major
                            const bf16_t* __restrict__ Xt,  // [B][P][512]
                            void* __restrict__ Out, int mode) {
  const v8f VZ = {0.f, 0.f, 0.f, 0.f, 0.f, 0.f, 0.f, 0.f};
  int lane = threadIdx.x & 31;
  int wave = threadIdx.x >> 5;
  int laneM = lane & 15, g = lane >> 4;
  int b  = blockIdx.z;
  int M0 = blockIdx.y * 128 + wave * 16;
  int N0 = blockIdx.x * 64;
  const bf16_t* Xb = Xt + (size_t)b * PDIM * CDIM;

  v8f acc[4];
#pragma unroll
  for (int t = 0; t < 4; ++t) acc[t] = VZ;

#if USE_ASYNC_LDS
  __shared__ __align__(16) bf16_t bpan[2][64 * 32];  // [col][K] panel, x2 buffers
  int pcol = threadIdx.x >> 2;            // 0..63  (N column)
  int pchk = (threadIdx.x & 3) * 8;       // 0,8,16,24 (K chunk, bf16 elems)
  const bf16_t* gB = Xb + (size_t)(N0 + pcol) * CDIM + pchk;
  cp16_async(gB, &bpan[0][pcol * 32 + pchk]);
  for (int kb = 0; kb < CDIM; kb += 32) {
    int cur = (kb >> 5) & 1;
    if (kb + 32 < CDIM) {
      cp16_async(gB + kb + 32, &bpan[cur ^ 1][pcol * 32 + pchk]);
      __builtin_amdgcn_s_wait_asynccnt(1);   // current panel done (in-order)
    } else {
      __builtin_amdgcn_s_wait_asynccnt(0);
    }
    __syncthreads();
    const bf16_t* wrow = W + (size_t)(M0 + laneM) * CDIM + kb;
    v16bf a = ld_frag(wrow + 16 * g, wrow + 16 * g + 8);
    const bf16_t* bp = &bpan[cur][0];
#pragma unroll
    for (int t = 0; t < 4; ++t) {
      const bf16_t* bc = bp + (16 * t + laneM) * 32 + 16 * g;
      v16bf bt = ld_frag(bc, bc + 8);
      acc[t] = wmma_bf16(a, bt, acc[t]);
    }
    __syncthreads();  // protect buffer before next overwrite
  }
#else
  for (int kb = 0; kb < CDIM; kb += 32) {
    const bf16_t* wrow = W + (size_t)(M0 + laneM) * CDIM + kb;
    v16bf a = ld_frag(wrow + 16 * g, wrow + 16 * g + 8);
#pragma unroll
    for (int t = 0; t < 4; ++t) {
      const bf16_t* xcol = Xb + (size_t)(N0 + 16 * t + laneM) * CDIM + kb + 16 * g;
      v16bf bt = ld_frag(xcol, xcol + 8);
      acc[t] = wmma_bf16(a, bt, acc[t]);
    }
  }
#endif

#pragma unroll
  for (int t = 0; t < 4; ++t) {
    int p = N0 + 16 * t + laneM;
#pragma unroll
    for (int i = 0; i < 8; ++i) {
      int o = M0 + i + 8 * g;      // C layout: vgpr i -> row i (+8 for upper half)
      float val = acc[t][i];
      if (mode == 2) {
        ((float*)Out)[((size_t)b * CDIM + o) * PDIM + p] = val;
      } else {
        int nh = o & 7, dh = o >> 3;
        int bn = b * NHH + nh;
        bf16_t* ob = (bf16_t*)Out;
        if (mode == 0) ob[((size_t)bn * PDIM + p) * DHH + dh] = (bf16_t)val;
        else           ob[((size_t)bn * DHH + dh) * PDIM + p] = (bf16_t)val;
      }
    }
  }
}

// ---------------------------------------------------------------------------
// 5) flash attention helpers
// ---------------------------------------------------------------------------
static __device__ __forceinline__ void online_softmax(v8f sc[2], float rm[8],
                                                      float rl[8], v8f acc[4]) {
  float fsc[8];
#pragma unroll
  for (int i = 0; i < 8; ++i) {
    float v0 = sc[0][i], v1 = sc[1][i];
    float mx = fmaxf(v0, v1);
#pragma unroll
    for (int msk = 1; msk < 16; msk <<= 1) mx = fmaxf(mx, __shfl_xor(mx, msk, 32));
    float mnew = fmaxf(rm[i], mx);
    float f  = __expf(rm[i] - mnew);
    float p0 = __expf(v0 - mnew), p1 = __expf(v1 - mnew);
    sc[0][i] = p0; sc[1][i] = p1;
    float ps = p0 + p1;
#pragma unroll
    for (int msk = 1; msk < 16; msk <<= 1) ps += __shfl_xor(ps, msk, 32);
    rl[i] = rl[i] * f + ps;
    rm[i] = mnew;
    fsc[i] = f;
  }
#pragma unroll
  for (int t = 0; t < 4; ++t)
#pragma unroll
    for (int i = 0; i < 8; ++i) acc[t][i] *= fsc[i];
}

// Transpose P tile (WMMA C layout -> A layout) through wave-private LDS.
static __device__ __forceinline__ v16bf p_to_afrag(const v8f sc[2], bf16_t* pb,
                                                   int laneM, int g) {
#pragma unroll
  for (int si = 0; si < 2; ++si)
#pragma unroll
    for (int i = 0; i < 8; ++i)
      pb[(i + 8 * g) * 32 + 16 * si + laneM] = (bf16_t)sc[si][i];
  asm volatile("" ::: "memory");  // same-wave LDS ops are in-order
  v16bf ap = ld_frag(pb + laneM * 32 + 16 * g, pb + laneM * 32 + 16 * g + 8);
  asm volatile("" ::: "memory");
  return ap;
}

// ---------------------------------------------------------------------------
// 5) flash attention: per wave a 16-row Q tile, stream 32-wide KV tiles.
//    K/V panels (shared by all 8 waves) double-buffered in LDS via async DMA.
// ---------------------------------------------------------------------------
__global__ void attn_kernel(const bf16_t* __restrict__ Q,  // [BN][P][DH]
                            const bf16_t* __restrict__ K,  // [BN][P][DH]
                            const bf16_t* __restrict__ V,  // [BN][DH][P]
                            bf16_t* __restrict__ AT) {     // [B][P][DE], c2=dh*8+nh
  __shared__ __align__(16) bf16_t pls[8][16 * 32];
  const v8f VZ = {0.f, 0.f, 0.f, 0.f, 0.f, 0.f, 0.f, 0.f};
  int lane = threadIdx.x & 31, wave = threadIdx.x >> 5;
  int laneM = lane & 15, g = lane >> 4;
  int bn = blockIdx.y;
  int b = bn >> 3, nh = bn & 7;
  int m0 = blockIdx.x * 128 + wave * 16;

  const bf16_t* qrow = Q + ((size_t)bn * PDIM + m0 + laneM) * DHH;
  v16bf aq0 = ld_frag(qrow + 16 * g, qrow + 16 * g + 8);        // K'=0..31
  v16bf aq1 = ld_frag(qrow + 32 + 16 * g, qrow + 40 + 16 * g);   // K'=32..63

  v8f acc[4];
#pragma unroll
  for (int t = 0; t < 4; ++t) acc[t] = VZ;
  float rm[8], rl[8];
#pragma unroll
  for (int i = 0; i < 8; ++i) { rm[i] = -3.0e38f; rl[i] = 0.f; }

  bf16_t* pb = &pls[wave][0];
  const bf16_t* Kb = K + (size_t)bn * PDIM * DHH;
  const bf16_t* Vb = V + (size_t)bn * DHH * PDIM;

#if USE_ASYNC_LDS
  __shared__ __align__(16) bf16_t kpan[2][32 * 64];  // [n][dh]
  __shared__ __align__(16) bf16_t vpan[2][64 * 32];  // [dh][n]
  int krow = threadIdx.x >> 3, kchk = (threadIdx.x & 7) * 8;  // 32 rows x 128B
  int vdh  = threadIdx.x >> 2, vchk = (threadIdx.x & 3) * 8;  // 64 rows x 64B
  const bf16_t* gK = Kb + (size_t)krow * DHH + kchk;
  const bf16_t* gV = Vb + (size_t)vdh * PDIM + vchk;
  cp16_async(gK, &kpan[0][krow * 64 + kchk]);
  cp16_async(gV, &vpan[0][vdh * 32 + vchk]);
  for (int n0 = 0; n0 < PDIM; n0 += 32) {
    int cur = (n0 >> 5) & 1;
    if (n0 + 32 < PDIM) {
      cp16_async(gK + (size_t)(n0 + 32) * DHH, &kpan[cur ^ 1][krow * 64 + kchk]);
      cp16_async(gV + (n0 + 32),               &vpan[cur ^ 1][vdh * 32 + vchk]);
      __builtin_amdgcn_s_wait_asynccnt(2);   // current K+V done (in-order)
    } else {
      __builtin_amdgcn_s_wait_asynccnt(0);
    }
    __syncthreads();
    v8f sc[2];
#pragma unroll
    for (int si = 0; si < 2; ++si) {
      const bf16_t* kr = &kpan[cur][(16 * si + laneM) * 64];
      v16bf bk0 = ld_frag(kr + 16 * g, kr + 16 * g + 8);
      v16bf bk1 = ld_frag(kr + 32 + 16 * g, kr + 40 + 16 * g);
      v8f c = VZ;
      c = wmma_bf16(aq0, bk0, c);
      c = wmma_bf16(aq1, bk1, c);
      sc[si] = c * 0.125f;  // 1/sqrt(64)
    }
    online_softmax(sc, rm, rl, acc);
    v16bf ap = p_to_afrag(sc, pb, laneM, g);
#pragma unroll
    for (int t = 0; t < 4; ++t) {
      const bf16_t* vc = &vpan[cur][(16 * t + laneM) * 32 + 16 * g];
      v16bf bv = ld_frag(vc, vc + 8);
      acc[t] = wmma_bf16(ap, bv, acc[t]);
    }
    __syncthreads();  // protect panels before next overwrite
  }
#else
  for (int n0 = 0; n0 < PDIM; n0 += 32) {
    if (n0 + 32 < PDIM) __builtin_prefetch(Kb + (size_t)(n0 + 32) * DHH, 0, 1);
    v8f sc[2];
#pragma unroll
    for (int si = 0; si < 2; ++si) {
      const bf16_t* krow2 = Kb + (size_t)(n0 + 16 * si + laneM) * DHH;
      v16bf bk0 = ld_frag(krow2 + 16 * g, krow2 + 16 * g + 8);
      v16bf bk1 = ld_frag(krow2 + 32 + 16 * g, krow2 + 40 + 16 * g);
      v8f c = VZ;
      c = wmma_bf16(aq0, bk0, c);
      c = wmma_bf16(aq1, bk1, c);
      sc[si] = c * 0.125f;
    }
    online_softmax(sc, rm, rl, acc);
    v16bf ap = p_to_afrag(sc, pb, laneM, g);
#pragma unroll
    for (int t = 0; t < 4; ++t) {
      const bf16_t* vcol = Vb + (size_t)(16 * t + laneM) * PDIM + n0 + 16 * g;
      v16bf bv = ld_frag(vcol, vcol + 8);
      acc[t] = wmma_bf16(ap, bv, acc[t]);
    }
  }
#endif

  // --- epilogue: normalize, emit [b][p][c2] with c2 = dh*8+nh ------------
#pragma unroll
  for (int t = 0; t < 4; ++t) {
    int dh = 16 * t + laneM;
#pragma unroll
    for (int i = 0; i < 8; ++i) {
      int m = m0 + i + 8 * g;
      float val = acc[t][i] / rl[i] + 1e-8f;
      AT[((size_t)b * PDIM + m) * (NHH * DHH) + dh * NHH + nh] = (bf16_t)val;
    }
  }
}

// ---------------------------------------------------------------------------
// launch
// ---------------------------------------------------------------------------
extern "C" void kernel_launch(void* const* d_in, const int* in_sizes, int n_in,
                              void* d_out, int out_size, void* d_ws, size_t ws_size,
                              hipStream_t stream) {
  (void)in_sizes; (void)n_in; (void)out_size; (void)ws_size;
  const float* x     = (const float*)d_in[0];
  const float* ctx   = (const float*)d_in[1];
  const float* gamma = (const float*)d_in[2];
  const float* wq    = (const float*)d_in[3];
  const float* wk    = (const float*)d_in[4];
  const float* wv    = (const float*)d_in[5];
  const float* wp    = (const float*)d_in[6];

  char* ws = (char*)d_ws;
  size_t off = 0;
  auto carve = [&](size_t bytes) {
    char* p = ws + off;
    off += (bytes + 255) & ~(size_t)255;
    return p;
  };
  const size_t TBF = (size_t)NB * PDIM * CDIM * sizeof(bf16_t);  // 8 MB
  bf16_t* xnT   = (bf16_t*)carve(TBF);
  bf16_t* cnT   = (bf16_t*)carve(TBF);
  bf16_t* q_ws  = (bf16_t*)carve(TBF);
  bf16_t* k_ws  = (bf16_t*)carve(TBF);
  bf16_t* v_ws  = (bf16_t*)carve(TBF);
  bf16_t* attnT = (bf16_t*)carve(TBF);
  bf16_t* wcat  = (bf16_t*)carve((size_t)4 * CDIM * CDIM * sizeof(bf16_t));
  float*  invx  = (float*)carve((size_t)NB * PDIM * sizeof(float));
  float*  invc  = (float*)carve((size_t)NB * PDIM * sizeof(float));
  bf16_t* wq_b = wcat;
  bf16_t* wk_b = wcat + (size_t)CDIM * CDIM;
  bf16_t* wv_b = wcat + (size_t)2 * CDIM * CDIM;
  bf16_t* wp_b = wcat + (size_t)3 * CDIM * CDIM;

  rownorm_kernel<<<64, 256, 0, stream>>>(x, ctx, invx, invc);
  normcast_kernel<<<dim3(32, 16, 16), dim3(32, 8, 1), 0, stream>>>(
      x, ctx, gamma, invx, invc, xnT, cnT);
  wcast_kernel<<<4096, 256, 0, stream>>>(wq, wk, wv, wp, wcat);

  gemm_kernel<<<dim3(16, 4, NB), 256, 0, stream>>>(wq_b, xnT, q_ws, 0);
  gemm_kernel<<<dim3(16, 4, NB), 256, 0, stream>>>(wk_b, cnT, k_ws, 0);
  gemm_kernel<<<dim3(16, 4, NB), 256, 0, stream>>>(wv_b, cnT, v_ws, 1);

  attn_kernel<<<dim3(8, 64), 256, 0, stream>>>(q_ws, k_ws, v_ws, attnT);

  gemm_kernel<<<dim3(16, 4, NB), 256, 0, stream>>>(wp_b, attnT, d_out, 2);
}
